// EnhancedLegalRGCN_33148557590803
// MI455X (gfx1250) — matrix operand order repaired and checked by reading, not compiled, not run on hardware
//
#include <hip/hip_runtime.h>
#include <stdint.h>

// ---------------------------------------------------------------------------
// EnhancedLegalRGCN for MI455X (gfx1250): bf16 WMMA GEMMs fed from LDS via
// async global->LDS copies (ASYNCcnt), f32 atomic scatter-mean aggregation.
// ---------------------------------------------------------------------------

typedef __attribute__((ext_vector_type(16))) __bf16 v16bf;
typedef __attribute__((ext_vector_type(8)))  __bf16 v8bf;
typedef __attribute__((ext_vector_type(4)))  __bf16 v4bf;
typedef __attribute__((ext_vector_type(8)))  float  v8f;

#define RGCN_N  50000
#define RGCN_E  800000
#define RGCN_IN 768
#define RGCN_H  128

__device__ __forceinline__ v8f wmma_bf16(v16bf a, v16bf b, v8f c) {
  return __builtin_amdgcn_wmma_f32_16x16x32_bf16(
      /*neg_a=*/false, a, /*neg_b=*/false, b,
      /*c_mod=*/(short)0, c, /*reuse_a=*/false, /*reuse_b=*/false);
}

// A fragment (16x32 bf16, row-major source): lane = M row (lane&15),
// K-halves: lanes 0-15 hold K 0..7 & 16..23, lanes 16-31 hold 8..15 & 24..31.
// Caller passes p already offset by (lane>>4)*8 within the K tile.
__device__ __forceinline__ v16bf load_afrag(const __bf16* p) {
  v8bf lo = *(const v8bf*)p;
  v8bf hi = *(const v8bf*)(p + 16);
  return __builtin_shufflevector(lo, hi, 0, 1, 2, 3, 4, 5, 6, 7,
                                 8, 9, 10, 11, 12, 13, 14, 15);
}

// Async 16-byte global -> LDS copy (tracked by ASYNCcnt).  lds_off is the
// workgroup-relative LDS byte offset (= low 32 bits of a flat shared ptr).
__device__ __forceinline__ void async_copy16(uint32_t lds_off,
                                             const void* gptr) {
  asm volatile("global_load_async_to_lds_b128 %0, %1, off"
               :: "v"(lds_off), "v"(gptr)
               : "memory");
}

__device__ __forceinline__ void wait_async0() {
  asm volatile("s_wait_asynccnt 0x0" ::: "memory");
}

__device__ __forceinline__ uint32_t lds_offset_of(const void* shared_ptr) {
  return (uint32_t)(uintptr_t)shared_ptr;  // flat LDS addr: low 32 bits
}

// --------------------------- small utility kernels -------------------------

__global__ void f32_to_bf16_kernel(const float* __restrict__ in,
                                   __bf16* __restrict__ out, long long n) {
  long long i = (long long)blockIdx.x * blockDim.x + threadIdx.x;
  if (i < n) out[i] = (__bf16)in[i];
}

__global__ void zero_f32_kernel(float* __restrict__ p, long long n) {
  long long i = (long long)blockIdx.x * blockDim.x + threadIdx.x;
  if (i < n) p[i] = 0.0f;
}

// Pack W[K, 512] = [root | rel0 | rel1 | rel2] into B-fragment layout:
// element index = ((kt*Mtiles + nt)*32 + lane)*16 + i,
// value = W[kt*32 + (lane>>4)*8 + (i&7) + (i>>3)*16][nt*16 + (lane&15)].
__global__ void pack_rgcn_kernel(const float* __restrict__ root,
                                 const float* __restrict__ rel,
                                 int K, __bf16* __restrict__ out) {
  int idx = blockIdx.x * blockDim.x + threadIdx.x;
  if (idx >= K * 512) return;
  int i    = idx & 15;
  int lane = (idx >> 4) & 31;
  int f    = idx >> 9;           // fragment stride = 512 elements
  int kt   = f >> 5;             // Mtiles = 512/16 = 32
  int nt   = f & 31;
  int n = nt * 16 + (lane & 15);
  int k = kt * 32 + (lane >> 4) * 8 + (i & 7) + (i >> 3) * 16;
  float v;
  if (n < 128) {
    v = root[(size_t)k * 128 + n];
  } else {
    int r = (n >> 7) - 1;
    v = rel[((size_t)r * K + k) * 128 + (n & 127)];
  }
  out[idx] = (__bf16)v;
}

__global__ void pack_plain_kernel(const float* __restrict__ W, int K, int M,
                                  __bf16* __restrict__ out) {
  int idx = blockIdx.x * blockDim.x + threadIdx.x;
  if (idx >= K * M) return;
  int i      = idx & 15;
  int lane   = (idx >> 4) & 31;
  int f      = idx >> 9;
  int Mtiles = M >> 4;
  int kt = f / Mtiles;
  int nt = f % Mtiles;
  int n = nt * 16 + (lane & 15);
  int k = kt * 32 + (lane >> 4) * 8 + (i & 7) + (i >> 3) * 16;
  out[idx] = (__bf16)W[(size_t)k * M + n];
}

__global__ void count_kernel(const int* __restrict__ ei,
                             const int* __restrict__ et,
                             float* __restrict__ cnt, int E, int Nn) {
  int e = blockIdx.x * blockDim.x + threadIdx.x;
  if (e < E) atomicAdd(&cnt[(size_t)et[e] * Nn + ei[(size_t)E + e]], 1.0f);
}

// --------------------------- WMMA GEMM -------------------------------------
// Y[nrows, M] (bf16) = A[nrows, K] (bf16) x Bpack (fragment layout).
// Block: 256 thr / 8 waves = 2 row-groups x 4 col-groups -> 64 rows x 256
// cols.  Per K-tile the block's 16KB B slice is async-staged into LDS
// (double buffered); each wave then runs 8 WMMAs (2 A frags x 4 B frags).

__global__ void __launch_bounds__(256)
gemm_bf16_kernel(const __bf16* __restrict__ A, const __bf16* __restrict__ Bp,
                 __bf16* __restrict__ Y, int nrows, int K, int M) {
  __shared__ __bf16 bstage[2][16 * 512];   // 2 x 16KB
  int tid  = threadIdx.x;
  int wave = tid >> 5;
  int lane = tid & 31;
  int rg  = wave >> 2;                     // row group 0..1
  int cgw = wave & 3;                      // col group 0..3 (64 cols each)
  int cb  = blockIdx.y;                    // 256-col block
  int Mtiles = M >> 4;

  int r0 = blockIdx.x * 64 + rg * 32;
  int r1 = r0 + 16;
  bool v0 = r0 < nrows;                    // nrows is a multiple of 16
  bool v1 = r1 < nrows;
  int rr0 = v0 ? r0 : 0;
  int rr1 = v1 ? r1 : 0;
  const __bf16* arow0 =
      A + (size_t)(rr0 + (lane & 15)) * K + ((lane >> 4) * 8);
  const __bf16* arow1 =
      A + (size_t)(rr1 + (lane & 15)) * K + ((lane >> 4) * 8);

  const char* bbytes = (const char*)Bp;
  uint32_t lds0 = lds_offset_of(&bstage[0][0]);
  uint32_t lds1 = lds_offset_of(&bstage[1][0]);

  auto issue_stage = [&](int kt, int buf) {
    size_t base = ((size_t)(kt * Mtiles + cb * 16) * 512) * 2;  // bytes
    uint32_t ldsb = buf ? lds1 : lds0;
#pragma unroll
    for (int q = 0; q < 4; ++q) {                // 256 thr * 4 * 16B = 16KB
      uint32_t off = (uint32_t)(tid + q * 256) * 16;
      async_copy16(ldsb + off, bbytes + base + off);
    }
  };

  v8f c[2][4] = {};
  int kts = K >> 5;
  issue_stage(0, 0);
  for (int kt = 0; kt < kts; ++kt) {
    wait_async0();          // this wave's stage-kt chunks are in LDS
    __syncthreads();        // everyone's chunks visible; prev buffer free
    if (kt + 1 < kts) issue_stage(kt + 1, (kt + 1) & 1);
    v16bf a0 = load_afrag(arow0 + kt * 32);
    v16bf a1 = load_afrag(arow1 + kt * 32);
    const __bf16* bb = &bstage[kt & 1][(size_t)(cgw * 4) * 512 + lane * 16];
#pragma unroll
    for (int j = 0; j < 4; ++j) {
      v16bf b = *(const v16bf*)(bb + (size_t)j * 512);
      c[0][j] = wmma_bf16(a0, b, c[0][j]);
      c[1][j] = wmma_bf16(a1, b, c[1][j]);
    }
  }

  int n0 = cb * 256 + cgw * 64 + (lane & 15);
  int m0 = rr0 + 8 * (lane >> 4);
  int m1 = rr1 + 8 * (lane >> 4);
  if (v0) {
#pragma unroll
    for (int v = 0; v < 8; ++v) {
      size_t row = (size_t)(m0 + v) * M;
#pragma unroll
      for (int j = 0; j < 4; ++j) Y[row + n0 + j * 16] = (__bf16)c[0][j][v];
    }
  }
  if (v1) {
#pragma unroll
    for (int v = 0; v < 8; ++v) {
      size_t row = (size_t)(m1 + v) * M;
#pragma unroll
      for (int j = 0; j < 4; ++j) Y[row + n0 + j * 16] = (__bf16)c[1][j][v];
    }
  }
}

// --------------------------- edge scatter / combine ------------------------
// One 32-lane group per edge; each lane handles 4 contiguous channels.

__global__ void scatter_kernel(const int* __restrict__ ei,
                               const int* __restrict__ et,
                               const __bf16* __restrict__ Y,
                               float* __restrict__ agg, int E, int Nn) {
  long long tid = (long long)blockIdx.x * blockDim.x + threadIdx.x;
  int e = (int)(tid >> 5);
  if (e >= E) return;
  int lane = (int)(tid & 31);
  int r = et[e], s = ei[e], d = ei[(size_t)E + e];
  const __bf16* h = Y + (size_t)s * 512 + 128 + r * 128 + lane * 4;
  v4bf hv = *(const v4bf*)h;
  float* a = agg + ((size_t)r * Nn + d) * 128 + lane * 4;
  atomicAdd(a + 0, (float)hv[0]);
  atomicAdd(a + 1, (float)hv[1]);
  atomicAdd(a + 2, (float)hv[2]);
  atomicAdd(a + 3, (float)hv[3]);
}

__global__ void combine_kernel(const __bf16* __restrict__ Y,
                               const float* __restrict__ agg,
                               const float* __restrict__ cnt,
                               const float* __restrict__ b,
                               __bf16* __restrict__ Xout, int Nn, int relu) {
  long long tid = (long long)blockIdx.x * blockDim.x + threadIdx.x;
  if (tid >= (long long)Nn * 128) return;
  int i = (int)(tid >> 7);
  int j = (int)(tid & 127);
  float v = (float)Y[(size_t)i * 512 + j] + b[j];
#pragma unroll
  for (int r = 0; r < 3; ++r) {
    float c = cnt[(size_t)r * Nn + i];
    v += agg[((size_t)r * Nn + i) * 128 + j] / fmaxf(c, 1.0f);
  }
  if (relu) v = fmaxf(v, 0.0f);
  Xout[tid] = (__bf16)v;
}

// --------------------------- edge head -------------------------------------
// One wave per 16-edge tile.  We1 (packed, 64KB) is async-staged into a 32KB
// LDS buffer in two halves (src K-tiles then dst K-tiles); the same buffer is
// then reused for the f32 hidden tile feeding the 128->3 VALU projection.

__global__ void __launch_bounds__(128)
edge_head_kernel(const int* __restrict__ ei, const __bf16* __restrict__ X,
                 const __bf16* __restrict__ We1p, const float* __restrict__ be1,
                 const float* __restrict__ We2, const float* __restrict__ be2,
                 float* __restrict__ out, int E) {
  __shared__ char smem[32768];
  __bf16* bst = (__bf16*)smem;
  int tid  = threadIdx.x;
  int wib  = tid >> 5;
  int lane = tid & 31;
  int t = blockIdx.x * 4 + wib;                 // 16-edge tile (grid exact)
  int e16 = t * 16 + (lane & 15);
  int s = ei[e16];
  int d = ei[(size_t)E + e16];
  int koff = (lane >> 4) * 8;
  const __bf16* srow = X + (size_t)s * 128 + koff;
  const __bf16* drow = X + (size_t)d * 128 + koff;
  const char* wbytes = (const char*)We1p;
  uint32_t ldsb = lds_offset_of(smem);

  // stage half 1: K-tiles 0..3 (src side), 32KB
#pragma unroll
  for (int q = 0; q < 16; ++q) {                // 128 thr * 16 * 16B = 32KB
    uint32_t off = (uint32_t)(tid + q * 128) * 16;
    async_copy16(ldsb + off, wbytes + off);
  }
  v8f c[8] = {};
  wait_async0();
  __syncthreads();
#pragma unroll
  for (int kt = 0; kt < 4; ++kt) {
    v16bf a = load_afrag(srow + kt * 32);
    const __bf16* bb = bst + (size_t)(kt * 8) * 512 + lane * 16;
#pragma unroll
    for (int nt = 0; nt < 8; ++nt) {
      v16bf b = *(const v16bf*)(bb + (size_t)nt * 512);
      c[nt] = wmma_bf16(a, b, c[nt]);
    }
  }
  __syncthreads();                              // half-1 reads done
  // stage half 2: K-tiles 4..7 (dst side)
#pragma unroll
  for (int q = 0; q < 16; ++q) {
    uint32_t off = (uint32_t)(tid + q * 128) * 16;
    async_copy16(ldsb + off, wbytes + 32768 + off);
  }
  wait_async0();
  __syncthreads();
#pragma unroll
  for (int kt = 0; kt < 4; ++kt) {
    v16bf a = load_afrag(drow + kt * 32);
    const __bf16* bb = bst + (size_t)(kt * 8) * 512 + lane * 16;
#pragma unroll
    for (int nt = 0; nt < 8; ++nt) {
      v16bf b = *(const v16bf*)(bb + (size_t)nt * 512);
      c[nt] = wmma_bf16(a, b, c[nt]);
    }
  }
  __syncthreads();                              // half-2 reads done: reuse smem
  float* hid = (float*)smem + (size_t)wib * 2048;   // 16x128 f32 per wave
  int mrow = 8 * (lane >> 4);
  int ncol = lane & 15;
#pragma unroll
  for (int nt = 0; nt < 8; ++nt) {
    float bv = be1[nt * 16 + ncol];
#pragma unroll
    for (int v = 0; v < 8; ++v) {
      float h = c[nt][v] + bv;
      hid[(mrow + v) * 128 + nt * 16 + ncol] = h > 0.0f ? h : 0.0f;
    }
  }
  // same-wave LDS ops are in order: safe to read our own region now
  if (lane < 16) {
    float a0 = be2[0], a1 = be2[1], a2 = be2[2];
    const float* hrow = &hid[lane * 128];
    for (int k = 0; k < 128; ++k) {
      float h = hrow[k];
      a0 = fmaf(h, We2[k * 3 + 0], a0);
      a1 = fmaf(h, We2[k * 3 + 1], a1);
      a2 = fmaf(h, We2[k * 3 + 2], a2);
    }
    size_t eo = (size_t)(t * 16 + lane) * 3;
    out[eo + 0] = a0;
    out[eo + 1] = a1;
    out[eo + 2] = a2;
  }
}

// --------------------------- node head -------------------------------------

__global__ void __launch_bounds__(128)
node_head_kernel(const __bf16* __restrict__ X, const __bf16* __restrict__ Wn1p,
                 const float* __restrict__ bn1, const float* __restrict__ Wn2,
                 const float* __restrict__ bn2, float* __restrict__ out,
                 int Nn) {
  __shared__ float lds[4][16 * 64];
  int wib  = threadIdx.x >> 5;
  int lane = threadIdx.x & 31;
  int t = blockIdx.x * 4 + wib;
  bool valid = (t * 16) < Nn;
  int tt = valid ? t : 0;                        // keep EXEC full / no early ret
  const __bf16* arow =
      X + (size_t)(tt * 16 + (lane & 15)) * 128 + (lane >> 4) * 8;
  v8f c[4] = {};
#pragma unroll
  for (int kt = 0; kt < 4; ++kt) {
    v16bf a = load_afrag(arow + kt * 32);
    const __bf16* bb = Wn1p + ((size_t)(kt * 4) * 32 + lane) * 16;
#pragma unroll
    for (int nt = 0; nt < 4; ++nt) {
      v16bf b = *(const v16bf*)(bb + (size_t)nt * 512);
      c[nt] = wmma_bf16(a, b, c[nt]);
    }
  }
  int mrow = 8 * (lane >> 4);
  int ncol = lane & 15;
#pragma unroll
  for (int nt = 0; nt < 4; ++nt) {
    float bv = bn1[nt * 16 + ncol];
#pragma unroll
    for (int v = 0; v < 8; ++v) {
      float h = c[nt][v] + bv;
      lds[wib][(mrow + v) * 64 + nt * 16 + ncol] = h > 0.0f ? h : 0.0f;
    }
  }
  __syncthreads();
  if (valid && lane < 16) {
    float a0 = bn2[0], a1 = bn2[1];
    const float* hrow = &lds[wib][lane * 64];
    for (int k = 0; k < 64; ++k) {
      float h = hrow[k];
      a0 = fmaf(h, Wn2[k * 2 + 0], a0);
      a1 = fmaf(h, Wn2[k * 2 + 1], a1);
    }
    size_t no = (size_t)(tt * 16 + lane) * 2;
    out[no + 0] = a0;
    out[no + 1] = a1;
  }
}

// --------------------------- launch ---------------------------------------

extern "C" void kernel_launch(void* const* d_in, const int* in_sizes, int n_in,
                              void* d_out, int out_size, void* d_ws,
                              size_t ws_size, hipStream_t stream) {
  (void)in_sizes; (void)n_in; (void)out_size; (void)ws_size;
  const int Nn = RGCN_N, Ee = RGCN_E, Kin = RGCN_IN;

  const float* x      = (const float*)d_in[0];
  const int*   ei     = (const int*)  d_in[1];   // [2, E]
  const int*   et     = (const int*)  d_in[2];   // [E]
  const float* Wrel1  = (const float*)d_in[3];
  const float* Wroot1 = (const float*)d_in[4];
  const float* b1     = (const float*)d_in[5];
  const float* Wrel2  = (const float*)d_in[6];
  const float* Wroot2 = (const float*)d_in[7];
  const float* b2     = (const float*)d_in[8];
  const float* Wrel3  = (const float*)d_in[9];
  const float* Wroot3 = (const float*)d_in[10];
  const float* b3     = (const float*)d_in[11];
  const float* We1    = (const float*)d_in[12];
  const float* be1    = (const float*)d_in[13];
  const float* We2    = (const float*)d_in[14];
  const float* be2    = (const float*)d_in[15];
  const float* Wn1    = (const float*)d_in[16];
  const float* bn1    = (const float*)d_in[17];
  const float* Wn2    = (const float*)d_in[18];
  const float* bn2    = (const float*)d_in[19];
  float* out = (float*)d_out;

  char* ws = (char*)d_ws;
  size_t off = 0;
  auto alloc = [&](size_t bytes) -> void* {
    void* p = ws + off;
    off = (off + bytes + 255) & ~(size_t)255;
    return p;
  };
  __bf16* xbf  = (__bf16*)alloc((size_t)Nn * Kin * 2);   // x in bf16
  __bf16* ybf  = (__bf16*)alloc((size_t)Nn * 512 * 2);   // [root|h0|h1|h2]
  float*  agg  = (float*) alloc((size_t)3 * Nn * 128 * 4);
  float*  cnt  = (float*) alloc((size_t)3 * Nn * 4);
  __bf16* xcur = (__bf16*)alloc((size_t)Nn * 128 * 2);   // layer activation
  __bf16* w1p  = (__bf16*)alloc((size_t)Kin * 512 * 2);
  __bf16* w2p  = (__bf16*)alloc((size_t)128 * 512 * 2);
  __bf16* w3p  = (__bf16*)alloc((size_t)128 * 512 * 2);
  __bf16* we1p = (__bf16*)alloc((size_t)256 * 128 * 2);
  __bf16* wn1p = (__bf16*)alloc((size_t)128 * 64 * 2);

  // --- prep: conversions, weight packing, edge counts ---
  {
    long long n = (long long)Nn * Kin;
    f32_to_bf16_kernel<<<(unsigned)((n + 255) / 256), 256, 0, stream>>>(x, xbf, n);
  }
  pack_rgcn_kernel<<<(Kin * 512 + 255) / 256, 256, 0, stream>>>(Wroot1, Wrel1, Kin, w1p);
  pack_rgcn_kernel<<<(128 * 512 + 255) / 256, 256, 0, stream>>>(Wroot2, Wrel2, 128, w2p);
  pack_rgcn_kernel<<<(128 * 512 + 255) / 256, 256, 0, stream>>>(Wroot3, Wrel3, 128, w3p);
  pack_plain_kernel<<<(256 * 128 + 255) / 256, 256, 0, stream>>>(We1, 256, 128, we1p);
  pack_plain_kernel<<<(128 * 64 + 255) / 256, 256, 0, stream>>>(Wn1, 128, 64, wn1p);
  zero_f32_kernel<<<(3 * Nn + 255) / 256, 256, 0, stream>>>(cnt, (long long)3 * Nn);
  count_kernel<<<(Ee + 255) / 256, 256, 0, stream>>>(ei, et, cnt, Ee, Nn);

  const long long aggN   = (long long)3 * Nn * 128;
  const unsigned  aggGrd = (unsigned)((aggN + 255) / 256);
  const unsigned  sctGrd = (unsigned)(((long long)Ee * 32 + 255) / 256);
  const long long cmbN   = (long long)Nn * 128;
  const unsigned  cmbGrd = (unsigned)((cmbN + 255) / 256);
  dim3 gemmGrd((Nn + 63) / 64, 512 / 256);

  // --- layer 1 (K = 768) ---
  gemm_bf16_kernel<<<gemmGrd, 256, 0, stream>>>(xbf, w1p, ybf, Nn, Kin, 512);
  zero_f32_kernel<<<aggGrd, 256, 0, stream>>>(agg, aggN);
  scatter_kernel<<<sctGrd, 256, 0, stream>>>(ei, et, ybf, agg, Ee, Nn);
  combine_kernel<<<cmbGrd, 256, 0, stream>>>(ybf, agg, cnt, b1, xcur, Nn, 1);

  // --- layer 2 (K = 128) ---
  gemm_bf16_kernel<<<gemmGrd, 256, 0, stream>>>(xcur, w2p, ybf, Nn, 128, 512);
  zero_f32_kernel<<<aggGrd, 256, 0, stream>>>(agg, aggN);
  scatter_kernel<<<sctGrd, 256, 0, stream>>>(ei, et, ybf, agg, Ee, Nn);
  combine_kernel<<<cmbGrd, 256, 0, stream>>>(ybf, agg, cnt, b2, xcur, Nn, 1);

  // --- layer 3 (K = 128, no ReLU) ---
  gemm_bf16_kernel<<<gemmGrd, 256, 0, stream>>>(xcur, w3p, ybf, Nn, 128, 512);
  zero_f32_kernel<<<aggGrd, 256, 0, stream>>>(agg, aggN);
  scatter_kernel<<<sctGrd, 256, 0, stream>>>(ei, et, ybf, agg, Ee, Nn);
  combine_kernel<<<cmbGrd, 256, 0, stream>>>(ybf, agg, cnt, b3, xcur, Nn, 0);

  // --- heads ---
  edge_head_kernel<<<Ee / 16 / 4, 128, 0, stream>>>(ei, xcur, we1p, be1, We2,
                                                    be2, out, Ee);
  node_head_kernel<<<(Nn / 16 + 3) / 4, 128, 0, stream>>>(
      xcur, wn1p, bn1, Wn2, bn2, out + (size_t)Ee * 3, Nn);
}
